// GGNNEncoder_33139967656249
// MI455X (gfx1250) — matrix operand order repaired and checked by reading, not compiled, not run on hardware
//
#include <hip/hip_runtime.h>

// GGNN encoder on gfx1250: T=12 B=32 N=1024 E=2 H=32 L=2 NST=3
#define T_ 12
#define B_ 32
#define N_ 1024
#define E_ 2
#define H_ 32
#define L_ 2
#define NST_ 3
#define K_ (E_ * N_)   // 2048, aggregation K dim

typedef __attribute__((ext_vector_type(16))) __bf16 bf16x16;
typedef __attribute__((ext_vector_type(8)))  __bf16 bf16x8;
typedef __attribute__((ext_vector_type(4)))  __bf16 bf16x4;
typedef __attribute__((ext_vector_type(8)))  float  f32x8;
typedef __attribute__((ext_vector_type(4)))  float  f32x4;
typedef __attribute__((ext_vector_type(4)))  int    i32x4;

#define WMMA(a, b, c) \
  __builtin_amdgcn_wmma_f32_16x16x32_bf16(false, (a), false, (b), (short)0, (c), false, false)

#define LDSB_STRIDE 264   // 256 K-elems per column + 8-elem (16B) pad: keeps b128 align, spreads banks

#if defined(__HIP_DEVICE_COMPILE__) && \
    __has_builtin(__builtin_amdgcn_global_load_async_to_lds_b128) && \
    __has_builtin(__builtin_amdgcn_s_wait_asynccnt)
#define HAVE_ASYNC 1
typedef __attribute__((address_space(1))) i32x4 as1_i32x4;   // global int4
typedef __attribute__((address_space(3))) i32x4 as3_i32x4;   // LDS int4
#else
#define HAVE_ASYNC 0
#endif

// ---------- operand builders (wave32, 16x16x32 bf16 WMMA layouts) ----------

// A operand: 16x32 bf16 tile from an f32 row-major tile (ld = row stride in elems).
// Layout: lanes 0-15 row m hold K=0..15; lanes 16-31 hold K=16..31 (matches B operand k order).
__device__ __forceinline__ bf16x16 loadA_f32(const float* rowbase, int ld) {
  int lane = threadIdx.x & 31;
  const float* p = rowbase + (lane & 15) * ld + ((lane >> 4) << 4);
  f32x8 x0 = *(const f32x8*)p;
  f32x8 x1 = *(const f32x8*)(p + 8);
  bf16x16 a;
#pragma unroll
  for (int i = 0; i < 8; ++i) { a[i] = (__bf16)x0[i]; a[i + 8] = (__bf16)x1[i]; }
  return a;
}

// A operand from bf16 row-major tile (global or LDS pointer).
__device__ __forceinline__ bf16x16 loadA_bf16(const __bf16* rowbase, int ld) {
  int lane = threadIdx.x & 31;
  const __bf16* p = rowbase + (lane & 15) * ld + ((lane >> 4) << 4);
  bf16x16 a;
  *(bf16x8*)&a          = *(const bf16x8*)p;          // K chunk 0
  *(((bf16x8*)&a) + 1)  = *(const bf16x8*)(p + 8);    // K chunk 1
  return a;
}

// B operand: 32x16 bf16 from k-contiguous (column-major) weight wT[col*32 + k].
// Contiguous k order used here (same as A): lane n (0-15) = column n holding K=0..15; lanes 16-31 hold K=16..31.
__device__ __forceinline__ bf16x16 loadB_w(const __bf16* wT, int half) {
  int lane = threadIdx.x & 31;
  return *(const bf16x16*)(wT + ((lane & 15) + half * 16) * 32 + ((lane >> 4) << 4));
}

// B operand from LDS panel lb[col * LDSB_STRIDE + k], 32-K slab kk within the staged 256-K super-tile.
__device__ __forceinline__ bf16x16 loadB_lds(const __bf16* lb, int half, int kk) {
  int lane = threadIdx.x & 31;
  return *(const bf16x16*)(lb + ((lane & 15) + half * 16) * LDSB_STRIDE + kk * 32 +
                           ((lane >> 4) << 4));
}

// ---------- async B-panel staging (256K x 32 cols, 16KB) ----------
// Each of the 256 threads moves 64B of one msgsT column; async path goes straight
// to LDS (ASYNCcnt), bypassing VGPRs and overlapping with WMMA on the other buffer.
__device__ __forceinline__ void stage_panel(const __bf16* mt, __bf16* lbuf, int ks) {
  int sf   = threadIdx.x >> 3;          // column 0..31
  int ksub = (threadIdx.x & 7) * 32;    // 32-elem sub-slab
  const __bf16* src = mt + (size_t)sf * K_ + ks + ksub;
  __bf16* dst = lbuf + sf * LDSB_STRIDE + ksub;
#if HAVE_ASYNC
#pragma unroll
  for (int j = 0; j < 4; ++j) {
    __builtin_amdgcn_global_load_async_to_lds_b128(
        (as1_i32x4*)(src + j * 8),
        (as3_i32x4*)(dst + j * 8), 0, 0);
  }
#else
  bf16x16 t0 = *(const bf16x16*)src;
  bf16x16 t1 = *(const bf16x16*)(src + 16);
  *(bf16x16*)dst        = t0;
  *(bf16x16*)(dst + 16) = t1;
#endif
}

__device__ __forceinline__ void wait_async0() {
#if HAVE_ASYNC
  __builtin_amdgcn_s_wait_asynccnt(0);
#endif
}

// ---------- conversion kernels ----------

// adjacency f32 -> bf16 (134MB result ~ resident in 192MB L2 across all 6 step-layers)
__global__ void k_cvt_adj(const float* __restrict__ src, __bf16* __restrict__ dst) {
  size_t i = ((size_t)blockIdx.x * 256 + threadIdx.x) * 4;
  f32x4 x = *(const f32x4*)(src + i);
  bf16x4 y;
#pragma unroll
  for (int j = 0; j < 4; ++j) y[j] = (__bf16)x[j];
  *(bf16x4*)(dst + i) = y;
}

// 32x32 weight mats: transpose + convert so B-operand reads are k-contiguous
__global__ void k_cvt_w(const float* __restrict__ src, __bf16* __restrict__ dst, int nmats) {
  int i = blockIdx.x * 256 + threadIdx.x;
  if (i >= nmats * 1024) return;
  int mat = i >> 10, idx = i & 1023;
  int k = idx & 31, col = idx >> 5;
  dst[mat * 1024 + col * 32 + k] = (__bf16)src[mat * 1024 + k * 32 + col];
}

// ---------- h = cur @ W_in[l]  (in-place safe: each wave reads only the rows it writes) ----------
__global__ void __launch_bounds__(256) k_proj(const float* cur, float* hout,
                                              const __bf16* __restrict__ wT) {
  int tile = blockIdx.x * 8 + (threadIdx.x >> 5);   // 16-row tile of the 393216-row matrix
  int lane = threadIdx.x & 31;
  bf16x16 a  = loadA_f32(cur + (size_t)tile * 16 * H_, H_);
  bf16x16 b0 = loadB_w(wT, 0), b1 = loadB_w(wT, 1);
  f32x8 acc0 = {}, acc1 = {};
  acc0 = WMMA(a, b0, acc0);
  acc1 = WMMA(a, b1, acc1);
  float* o = hout + (size_t)tile * 16 * H_;
  int f = lane & 15, Mb = (lane >> 4) << 3;
#pragma unroll
  for (int v = 0; v < 8; ++v) {
    o[(Mb + v) * H_ + f]      = acc0[v];
    o[(Mb + v) * H_ + f + 16] = acc1[v];
  }
}

// ---------- msgsT[t][b][f][e*N+n] = (h @ W_edge[l][e])^T  (transposed store is a packed b128) ----------
__global__ void __launch_bounds__(256) k_msgs(const float* __restrict__ h,
                                              __bf16* __restrict__ msT,
                                              const __bf16* __restrict__ wTe) {
  int tile = blockIdx.x * 8 + (threadIdx.x >> 5);
  int lane = threadIdx.x & 31;
  int tb   = tile >> 6;             // (t*B+b); 64 node-tiles per (t,b)
  int nloc = (tile & 63) * 16;
  bf16x16 a = loadA_f32(h + (size_t)tile * 16 * H_, H_);
  __bf16* mt = msT + (size_t)tb * (H_ * K_);
  int f = lane & 15, Mb = (lane >> 4) << 3;
#pragma unroll
  for (int e = 0; e < E_; ++e) {
    bf16x16 b0 = loadB_w(wTe + e * 1024, 0);
    bf16x16 b1 = loadB_w(wTe + e * 1024, 1);
    f32x8 acc0 = {}, acc1 = {};
    acc0 = WMMA(a, b0, acc0);
    acc1 = WMMA(a, b1, acc1);
    bf16x8 p0, p1;
#pragma unroll
    for (int v = 0; v < 8; ++v) { p0[v] = (__bf16)acc0[v]; p1[v] = (__bf16)acc1[v]; }
    // accumulator lane holds consecutive M (= node) values for a fixed column f -> contiguous in msgsT
    *(bf16x8*)(mt + (size_t)f * K_        + e * N_ + nloc + Mb) = p0;
    *(bf16x8*)(mt + (size_t)(f + 16) * K_ + e * N_ + nloc + Mb) = p1;
  }
}

// ---------- fused: m = adj[b] @ msgs[t,b]  then GRU update of h (in place) ----------
__global__ void __launch_bounds__(256) k_agg_gru(
    const __bf16* __restrict__ adjB, const __bf16* __restrict__ msT,
    float* h, const __bf16* __restrict__ wT, int l,
    const float* __restrict__ bz, const float* __restrict__ br,
    const float* __restrict__ bh) {
  int blk  = blockIdx.x;                 // T*B*8 blocks
  int tb   = blk >> 3;                   // t*B + b
  int b    = tb % B_;
  int wave = threadIdx.x >> 5, lane = threadIdx.x & 31;
  int mtile = (blk & 7) * 8 + wave;      // 0..63: 16-node output tile
  int f = lane & 15, Mb = (lane >> 4) << 3;
  int c0 = (lane >> 4) << 4;             // A-operand K chunk base

  __shared__ __align__(16) __bf16 lb[2][32 * LDSB_STRIDE];  // double-buffered B panel
  __shared__ __align__(16) __bf16 wtile[8][16 * H_];        // per-wave relayout scratch

  const __bf16* arow = adjB + ((size_t)b * N_ + mtile * 16 + (lane & 15)) * K_;
  const __bf16* mt   = msT + (size_t)tb * (H_ * K_);

  f32x8 acc0 = {}, acc1 = {};
  stage_panel(mt, lb[0], 0);
  for (int s = 0; s < 8; ++s) {          // 8 super-tiles of 256 K
    wait_async0();
    __syncthreads();                     // panel s resident; all waves done with panel s-1
    if (s < 7) stage_panel(mt, lb[(s + 1) & 1], (s + 1) * 256);  // prefetch into vacated buffer
    const __bf16* cur = lb[s & 1];
    int ks = s * 256;
#pragma unroll
    for (int kk = 0; kk < 8; ++kk) {
      int kg = ks + kk * 32;
      bf16x16 a;
      *(bf16x8*)&a         = *(const bf16x8*)(arow + kg + c0);
      *(((bf16x8*)&a) + 1) = *(const bf16x8*)(arow + kg + c0 + 8);
      bf16x16 b0 = loadB_lds(cur, 0, kk);
      bf16x16 b1 = loadB_lds(cur, 1, kk);
      acc0 = WMMA(a, b0, acc0);
      acc1 = WMMA(a, b1, acc1);
    }
  }

  // ---- GRU: m only ever feeds GEMMs, so re-layout the 16x32 accum as an A operand via LDS ----
  __bf16* tile = &wtile[wave][0];
#pragma unroll
  for (int v = 0; v < 8; ++v) {
    tile[(Mb + v) * H_ + f]      = (__bf16)acc0[v];
    tile[(Mb + v) * H_ + f + 16] = (__bf16)acc1[v];
  }
  __syncthreads();

  bf16x16 Am = loadA_bf16(tile, H_);
  float* hrow = h + ((size_t)tb * N_ + mtile * 16) * H_;
  bf16x16 Ah = loadA_f32(hrow, H_);

  const __bf16* WzT = wT + (6  + l) * 1024;
  const __bf16* UzT = wT + (8  + l) * 1024;
  const __bf16* WrT = wT + (10 + l) * 1024;
  const __bf16* UrT = wT + (12 + l) * 1024;
  const __bf16* WhT = wT + (14 + l) * 1024;
  const __bf16* UhT = wT + (16 + l) * 1024;

  f32x8 z0 = {}, z1 = {}, r0 = {}, r1 = {};
  z0 = WMMA(Am, loadB_w(WzT, 0), z0);  z0 = WMMA(Ah, loadB_w(UzT, 0), z0);
  z1 = WMMA(Am, loadB_w(WzT, 1), z1);  z1 = WMMA(Ah, loadB_w(UzT, 1), z1);
  r0 = WMMA(Am, loadB_w(WrT, 0), r0);  r0 = WMMA(Ah, loadB_w(UrT, 0), r0);
  r1 = WMMA(Am, loadB_w(WrT, 1), r1);  r1 = WMMA(Ah, loadB_w(UrT, 1), r1);

  float bz0 = bz[f], bz1 = bz[f + 16];
  float br0 = br[f], br1 = br[f + 16];
  float bh0 = bh[f], bh1 = bh[f + 16];

  float hv0[8], hv1[8], zv0[8], zv1[8], rv0[8], rv1[8];
#pragma unroll
  for (int v = 0; v < 8; ++v) {
    hv0[v] = hrow[(Mb + v) * H_ + f];
    hv1[v] = hrow[(Mb + v) * H_ + f + 16];
    zv0[v] = 1.f / (1.f + __expf(-(z0[v] + bz0)));
    zv1[v] = 1.f / (1.f + __expf(-(z1[v] + bz1)));
    rv0[v] = 1.f / (1.f + __expf(-(r0[v] + br0)));
    rv1[v] = 1.f / (1.f + __expf(-(r1[v] + br1)));
  }
  __syncthreads();
#pragma unroll
  for (int v = 0; v < 8; ++v) {   // (r*h) as next A operand
    tile[(Mb + v) * H_ + f]      = (__bf16)(rv0[v] * hv0[v]);
    tile[(Mb + v) * H_ + f + 16] = (__bf16)(rv1[v] * hv1[v]);
  }
  __syncthreads();
  bf16x16 Arh = loadA_bf16(tile, H_);

  f32x8 hc0a = {}, hc1a = {};
  hc0a = WMMA(Am, loadB_w(WhT, 0), hc0a);  hc0a = WMMA(Arh, loadB_w(UhT, 0), hc0a);
  hc1a = WMMA(Am, loadB_w(WhT, 1), hc1a);  hc1a = WMMA(Arh, loadB_w(UhT, 1), hc1a);

#pragma unroll
  for (int v = 0; v < 8; ++v) {
    float hc0 = tanhf(hc0a[v] + bh0);
    float hc1 = tanhf(hc1a[v] + bh1);
    hrow[(Mb + v) * H_ + f]      = (1.f - zv0[v]) * hv0[v] + zv0[v] * hc0;
    hrow[(Mb + v) * H_ + f + 16] = (1.f - zv1[v]) * hv1[v] + zv1[v] * hc1;
  }
}

// ---------- host driver ----------
// workspace layout (bytes):
//   adjB  @ 0          : 134,217,728  (B*N*E*N bf16)
//   wT    @ 134217728  :      36,864  (18 transposed 32x32 bf16 mats)
//   h     @ 134254592  :  50,331,648  (T*B*N*H f32)
//   msT   @ 184586240  :  50,331,648  (T*B*H*E*N bf16)   -> total ~224 MB
extern "C" void kernel_launch(void* const* d_in, const int* in_sizes, int n_in,
                              void* d_out, int out_size, void* d_ws, size_t ws_size,
                              hipStream_t stream) {
  (void)in_sizes; (void)n_in; (void)out_size; (void)ws_size;
  const float* inp = (const float*)d_in[0];
  const float* adj = (const float*)d_in[2];
  const float* Win = (const float*)d_in[3];
  const float* Wed = (const float*)d_in[4];
  const float* Wz  = (const float*)d_in[5];
  const float* Uz  = (const float*)d_in[6];
  const float* Wr  = (const float*)d_in[7];
  const float* Ur  = (const float*)d_in[8];
  const float* Wh  = (const float*)d_in[9];
  const float* Uh  = (const float*)d_in[10];
  const float* bz  = (const float*)d_in[11];
  const float* br  = (const float*)d_in[12];
  const float* bh  = (const float*)d_in[13];
  float* out = (float*)d_out;

  char* ws = (char*)d_ws;
  __bf16* adjB = (__bf16*)(ws);
  __bf16* wT   = (__bf16*)(ws + 134217728);
  float*  h    = (float*) (ws + 134254592);
  __bf16* msT  = (__bf16*)(ws + 184586240);

  const size_t BNH = (size_t)B_ * N_ * H_;

  k_cvt_adj<<<65536, 256, 0, stream>>>(adj, adjB);
  k_cvt_w<<<8,  256, 0, stream>>>(Win, wT + 0  * 1024, 2);
  k_cvt_w<<<16, 256, 0, stream>>>(Wed, wT + 2  * 1024, 4);
  k_cvt_w<<<8,  256, 0, stream>>>(Wz,  wT + 6  * 1024, 2);
  k_cvt_w<<<8,  256, 0, stream>>>(Uz,  wT + 8  * 1024, 2);
  k_cvt_w<<<8,  256, 0, stream>>>(Wr,  wT + 10 * 1024, 2);
  k_cvt_w<<<8,  256, 0, stream>>>(Ur,  wT + 12 * 1024, 2);
  k_cvt_w<<<8,  256, 0, stream>>>(Wh,  wT + 14 * 1024, 2);
  k_cvt_w<<<8,  256, 0, stream>>>(Uh,  wT + 16 * 1024, 2);

  for (int l = 0; l < L_; ++l) {
    k_proj<<<3072, 256, 0, stream>>>(l == 0 ? inp : (const float*)h, h, wT + l * 1024);
    for (int s = 0; s < NST_; ++s) {
      k_msgs<<<3072, 256, 0, stream>>>(h, msT, wT + (2 + l * 2) * 1024);
      k_agg_gru<<<T_ * B_ * 8, 256, 0, stream>>>(adjB, msT, h, wT, l,
                                                 bz + l * H_, br + l * H_, bh + l * H_);
    }
    // output_hidden[l] = h at last timestep
    (void)hipMemcpyAsync(out + (size_t)l * BNH, h + (size_t)(T_ - 1) * BNH,
                         BNH * sizeof(float), hipMemcpyDeviceToDevice, stream);
  }
  // outputs = final layer h over all timesteps
  (void)hipMemcpyAsync(out + (size_t)L_ * BNH, h, (size_t)T_ * BNH * sizeof(float),
                       hipMemcpyDeviceToDevice, stream);
}
